// MemoryPredictorSSM_33741263077640
// MI455X (gfx1250) — compile-verified, matrix operand order checked by reading, not run on hardware
//
#include <hip/hip_runtime.h>
#include <hip/hip_bf16.h>
#include <math.h>

// Problem constants (from reference)
#define BATCH 8
#define SEQ   1024
#define FEAT  64
#define DIM   256
#define SDIM  32
#define KCONV 4
#define NHEAD 5
#define DHEAD 128
#define BL    (BATCH * SEQ)          // 8192 rows
#define LN_EPSF 1e-5f

typedef __attribute__((ext_vector_type(16))) __bf16 v16bf;
typedef __attribute__((ext_vector_type(8)))  float  v8f;

__device__ __forceinline__ float warp_sum(float v) {
#pragma unroll
  for (int off = 16; off; off >>= 1) v += __shfl_xor(v, off, 32);
  return v;
}

// ACT: 0=none, 1=softplus, 2=sigmoid, 3=exact gelu
template <int ACT>
__device__ __forceinline__ float act_apply(float v) {
  if (ACT == 1) return (v > 20.f) ? v : log1pf(__expf(v));
  if (ACT == 2) return 1.f / (1.f + __expf(-v));
  if (ACT == 3) return 0.5f * v * (1.f + erff(v * 0.7071067811865475f));
  return v;
}

// ---------------------------------------------------------------------------
// Weight pre-pack: W[K,N] (f32, row-major, ld=N) -> bf16 in WMMA B-fragment
// order: [n_tile][k_step][lane][16 elems]. One thread per (n_tile,k_step,lane)
// writes 16 contiguous bf16 (32 B). Runs once per launch; weights are tiny
// and L2-resident, so this converts the per-tile 16 strided scalar loads in
// the GEMM hot loop into a single aligned 32-byte vector load.
// ---------------------------------------------------------------------------
__global__ void pack_weight_kernel(const float* __restrict__ W,
                                   __bf16* __restrict__ P, int K, int N) {
  const int tid   = blockIdx.x * blockDim.x + threadIdx.x;
  const int total = (N / 16) * (K / 32) * 32;
  if (tid >= total) return;
  const int lane = tid & 31;
  const int kt   = (tid >> 5) % (K / 32);
  const int nt   = (tid >> 5) / (K / 32);
  const int hi   = lane >> 4;
  const int lml  = lane & 15;
  __bf16* dst = P + (size_t)tid * 16;
#pragma unroll
  for (int e = 0; e < 16; ++e) {
    const int k = kt * 32 + hi * 8 + (e & 7) + ((e >> 3) << 4);
    dst[e] = (__bf16)W[(size_t)k * N + nt * 16 + lml];
  }
}

// ---------------------------------------------------------------------------
// Dense GEMM: C[M,N] = ACT( A[M,K](f32) * B(pre-packed bf16) + bias + resid )
// bf16 WMMA 16x16x32, f32 accumulate.
// One wave -> 64x16 output tile (4 accumulators); one B fragment (single
// 32-byte load) is reused across 4 A fragments per K-step; A loaded as b128.
// blockDim = 128 (4 waves => 256 rows of M per block), grid = (N/16, M/256).
// Fragment layouts per CDNA5 ISA 7.12.2 (16-bit A 16x32 / B 32x16, f32 C/D).
// ---------------------------------------------------------------------------
template <int ACT>
__global__ void wmma_gemm_bf16(const float* __restrict__ A, int lda,
                               const __bf16* __restrict__ Bp,
                               const float* __restrict__ bias,
                               const float* __restrict__ resid,
                               float* __restrict__ Cout, int ldc,
                               int M, int K) {
  const int lane = threadIdx.x & 31;
  const int wib  = threadIdx.x >> 5;                 // wave in block (0..3)
  const int m0   = (blockIdx.y * 4 + wib) * 64;      // 4 x 16 M-tiles per wave
  if (m0 >= M) return;
  const int n0   = blockIdx.x * 16;

  const int hi  = lane >> 4;                          // lane half (0/1)
  const int lml = lane & 15;                          // row (A) / col (B)

  // Pre-packed B fragments for this n-tile: stride 512 bf16 per K-step.
  const __bf16* Bt = Bp + (size_t)blockIdx.x * (size_t)K * 16 + (size_t)lane * 16;

  v8f acc[4] = {{}, {}, {}, {}};

  for (int k0 = 0; k0 < K; k0 += 32, Bt += 512) {
    const v16bf bfr = *(const v16bf*)Bt;              // 32 B aligned vector load
#pragma unroll
    for (int mt = 0; mt < 4; ++mt) {
      // A fragment: per lane 2x8 consecutive floats -> 4 x b128 loads.
      const float* Ar = A + (size_t)(m0 + mt * 16 + lml) * lda + k0 + hi * 8;
      const float4 a0 = *(const float4*)(Ar + 0);
      const float4 a1 = *(const float4*)(Ar + 4);
      const float4 a2 = *(const float4*)(Ar + 16);
      const float4 a3 = *(const float4*)(Ar + 20);
      v16bf afr;
      afr[0]  = (__bf16)a0.x; afr[1]  = (__bf16)a0.y;
      afr[2]  = (__bf16)a0.z; afr[3]  = (__bf16)a0.w;
      afr[4]  = (__bf16)a1.x; afr[5]  = (__bf16)a1.y;
      afr[6]  = (__bf16)a1.z; afr[7]  = (__bf16)a1.w;
      afr[8]  = (__bf16)a2.x; afr[9]  = (__bf16)a2.y;
      afr[10] = (__bf16)a2.z; afr[11] = (__bf16)a2.w;
      afr[12] = (__bf16)a3.x; afr[13] = (__bf16)a3.y;
      afr[14] = (__bf16)a3.z; afr[15] = (__bf16)a3.w;
      acc[mt] = __builtin_amdgcn_wmma_f32_16x16x32_bf16(
          /*neg_a=*/false, afr, /*neg_b=*/false, bfr,
          /*c_mod=*/(short)0, acc[mt], /*reuse_a=*/false, /*reuse_b=*/false);
    }
  }

  const int n = n0 + lml;
  const float bv = bias ? bias[n] : 0.f;
#pragma unroll
  for (int mt = 0; mt < 4; ++mt) {
#pragma unroll
    for (int r = 0; r < 8; ++r) {
      const int m = m0 + mt * 16 + r + hi * 8;        // C/D: lanes16-31 -> M+8
      float v = acc[mt][r] + bv;
      if (resid) v += resid[(size_t)m * ldc + n];
      Cout[(size_t)m * ldc + n] = act_apply<ACT>(v);
    }
  }
}

// ---------------------------------------------------------------------------
// LayerNorm over last dim (DIM=256). One wave per row, 8 values per lane.
// ---------------------------------------------------------------------------
__global__ void layernorm_kernel(const float* __restrict__ x,
                                 const float* __restrict__ g,
                                 const float* __restrict__ b,
                                 float* __restrict__ y, int rows) {
  const int row  = blockIdx.x * (blockDim.x >> 5) + (threadIdx.x >> 5);
  const int lane = threadIdx.x & 31;
  if (row >= rows) return;
  const float* xr = x + (size_t)row * DIM;
  float v[8], s = 0.f;
#pragma unroll
  for (int j = 0; j < 8; ++j) { v[j] = xr[lane + j * 32]; s += v[j]; }
  const float mean = warp_sum(s) * (1.f / DIM);
  float q = 0.f;
#pragma unroll
  for (int j = 0; j < 8; ++j) { const float d = v[j] - mean; q += d * d; }
  const float rstd = rsqrtf(warp_sum(q) * (1.f / DIM) + LN_EPSF);
  float* yr = y + (size_t)row * DIM;
#pragma unroll
  for (int j = 0; j < 8; ++j) {
    const int c = lane + j * 32;
    yr[c] = (v[j] - mean) * rstd * g[c] + b[c];
  }
}

// ---------------------------------------------------------------------------
// Depthwise causal conv (K=4, left pad 3) fused with SiLU. Elementwise grid.
// ---------------------------------------------------------------------------
__global__ void conv_silu_kernel(const float* __restrict__ x,
                                 const float* __restrict__ w,
                                 const float* __restrict__ cb,
                                 float* __restrict__ y) {
  const int idx = blockIdx.x * blockDim.x + threadIdx.x;
  if (idx >= BATCH * SEQ * DIM) return;
  const int d = idx % DIM;
  const int t = (idx / DIM) % SEQ;
  const int b = idx / (DIM * SEQ);
  const float* xb = x + (size_t)b * SEQ * DIM;
  float acc = cb[d];
#pragma unroll
  for (int j = 0; j < KCONV; ++j) {
    const int tt = t - (KCONV - 1) + j;
    if (tt >= 0) acc += w[d * KCONV + j] * xb[(size_t)tt * DIM + d];
  }
  y[idx] = acc / (1.f + __expf(-acc));                // silu
}

// ---------------------------------------------------------------------------
// Selective scan: one wave per (b,d) channel, lane = state index s (S=32).
// h_s <- exp(A[d,s]*dt) * h_s + dt*x*B[t,s];  y = sum_s h_s*C[t,s] (shfl_xor)
// Gate multiply fused into the output write. 2048 waves, no LDS, no barriers.
// ---------------------------------------------------------------------------
__global__ void scan_kernel(const float* __restrict__ delta,
                            const float* __restrict__ xc,
                            const float* __restrict__ Bm,
                            const float* __restrict__ Cm,
                            const float* __restrict__ log_a,
                            const float* __restrict__ gate,
                            float* __restrict__ yg) {
  const int wave = (blockIdx.x * blockDim.x + threadIdx.x) >> 5;
  const int lane = threadIdx.x & 31;
  if (wave >= BATCH * DIM) return;
  const int b = wave >> 8;                            // / DIM
  const int d = wave & (DIM - 1);

  const float a = -__expf(log_a[d * SDIM + lane]);    // A[d,s]
  float h = 0.f;

  const float* dp = delta + (size_t)b * SEQ * DIM + d;
  const float* xp = xc    + (size_t)b * SEQ * DIM + d;
  const float* gp = gate  + (size_t)b * SEQ * DIM + d;
  const float* bp = Bm    + (size_t)b * SEQ * SDIM + lane;
  const float* cp = Cm    + (size_t)b * SEQ * SDIM + lane;
  float*       yp = yg    + (size_t)b * SEQ * DIM + d;

  for (int t = 0; t < SEQ; ++t) {
    const float dt = dp[(size_t)t * DIM];             // broadcast across lanes
    const float xt = xp[(size_t)t * DIM];
    const float bt = bp[(size_t)t * SDIM];
    const float ct = cp[(size_t)t * SDIM];
    h = __expf(a * dt) * h + dt * xt * bt;
    const float y = warp_sum(h * ct);
    if (lane == 0) yp[(size_t)t * DIM] = y * gp[(size_t)t * DIM];
  }
}

// ---------------------------------------------------------------------------
// Final head dot: out[row] = dot(t[row,:DHEAD], w2[hh]) + b2[hh].
// One wave per row, 4 elements per lane.
// ---------------------------------------------------------------------------
__global__ void head_dot_kernel(const float* __restrict__ t,
                                const float* __restrict__ w2,
                                const float* __restrict__ b2,
                                float* __restrict__ out, int hh) {
  const int row  = blockIdx.x * (blockDim.x >> 5) + (threadIdx.x >> 5);
  const int lane = threadIdx.x & 31;
  if (row >= BL) return;
  const float* tr = t + (size_t)row * DHEAD;
  const float* wr = w2 + hh * DHEAD;
  float s = 0.f;
#pragma unroll
  for (int j = 0; j < DHEAD / 32; ++j) s += tr[lane + 32 * j] * wr[lane + 32 * j];
  s = warp_sum(s);
  if (lane == 0) out[row] = s + b2[hh];
}

// ---------------------------------------------------------------------------
extern "C" void kernel_launch(void* const* d_in, const int* in_sizes, int n_in,
                              void* d_out, int out_size, void* d_ws, size_t ws_size,
                              hipStream_t stream) {
  (void)in_sizes; (void)n_in; (void)out_size; (void)ws_size;

  // Flattened setup_inputs() order
  const float* x       = (const float*)d_in[0];
  const float* embed_w = (const float*)d_in[1];
  const float* embed_b = (const float*)d_in[2];
  const float* nin_g   = (const float*)d_in[3];
  const float* nin_b   = (const float*)d_in[4];
  const float* n1_g    = (const float*)d_in[5];
  const float* n1_b    = (const float*)d_in[6];
  const float* n2_g    = (const float*)d_in[7];
  const float* n2_b    = (const float*)d_in[8];
  // ssm1: d_in[9..19], ssm2: d_in[20..30]  (conv_w, conv_b, Wd, bd, Wb, Wc, log_a, Wg, bg, Wo, bo)
  const float* head_w1 = (const float*)d_in[31];
  const float* head_b1 = (const float*)d_in[32];
  const float* head_w2 = (const float*)d_in[33];
  const float* head_b2 = (const float*)d_in[34];
  float* out = (float*)d_out;

  // Workspace carve-out
  const size_t BLD = (size_t)BL * DIM;
  const size_t BLS = (size_t)BL * SDIM;
  float* ws    = (float*)d_ws;
  float* h     = ws; ws += BLD;     // hidden state (ping)
  float* h2    = ws; ws += BLD;     // hidden state (pong)
  float* tmp   = ws; ws += BLD;     // LN output / pre-LN embed
  float* xcv   = ws; ws += BLD;     // conv+silu output
  float* delta = ws; ws += BLD;
  float* gate  = ws; ws += BLD;
  float* yg    = ws; ws += BLD;     // y * gate
  float* Bmb   = ws; ws += BLS;
  float* Cmb   = ws; ws += BLS;
  float* thead = ws; ws += (size_t)BL * DHEAD;

  // Packed bf16 weights (fragment order); 32B-aligned since all float
  // carve-outs above are multiples of 64 elements.
  __bf16* wp = (__bf16*)ws;
  __bf16* p_embed = wp; wp += (size_t)FEAT * DIM;
  __bf16* p_Wd[2], *p_Wb[2], *p_Wc[2], *p_Wg[2], *p_Wo[2];
  for (int blk = 0; blk < 2; ++blk) {
    p_Wd[blk] = wp; wp += (size_t)DIM * DIM;
    p_Wb[blk] = wp; wp += (size_t)DIM * SDIM;
    p_Wc[blk] = wp; wp += (size_t)DIM * SDIM;
    p_Wg[blk] = wp; wp += (size_t)DIM * DIM;
    p_Wo[blk] = wp; wp += (size_t)DIM * DIM;
  }
  __bf16* p_head = wp; wp += (size_t)NHEAD * DIM * DHEAD;

  const dim3 gBlk(128);
  const dim3 gGridD(DIM / 16, BL / 256);    // N=256 GEMMs
  const dim3 gGridS(SDIM / 16, BL / 256);   // N=32 GEMMs
  const dim3 gGridH(DHEAD / 16, BL / 256);  // N=128 GEMMs
  const int elemThreads = 256;

  // 0) pack all weights into WMMA B-fragment order (bf16)
  {
    auto pack = [&](const float* W, __bf16* P, int K, int N) {
      const int total = (N / 16) * (K / 32) * 32;
      pack_weight_kernel<<<(total + 255) / 256, 256, 0, stream>>>(W, P, K, N);
    };
    pack(embed_w, p_embed, FEAT, DIM);
    for (int blk = 0; blk < 2; ++blk) {
      const int base = blk ? 20 : 9;
      pack((const float*)d_in[base + 2], p_Wd[blk], DIM, DIM);
      pack((const float*)d_in[base + 4], p_Wb[blk], DIM, SDIM);
      pack((const float*)d_in[base + 5], p_Wc[blk], DIM, SDIM);
      pack((const float*)d_in[base + 7], p_Wg[blk], DIM, DIM);
      pack((const float*)d_in[base + 9], p_Wo[blk], DIM, DIM);
    }
    for (int hh = 0; hh < NHEAD; ++hh)
      pack(head_w1 + (size_t)hh * DIM * DHEAD, p_head + (size_t)hh * DIM * DHEAD,
           DIM, DHEAD);
  }

  // 1) embed GEMM (M=8192, K=64, N=256) -> tmp, then LN -> h
  wmma_gemm_bf16<0><<<gGridD, gBlk, 0, stream>>>(
      x, FEAT, p_embed, embed_b, nullptr, tmp, DIM, BL, FEAT);
  layernorm_kernel<<<BL / 8, elemThreads, 0, stream>>>(tmp, nin_g, nin_b, h, BL);

  // 2) two SSM blocks (ping-pong h <-> h2)
  for (int blk = 0; blk < 2; ++blk) {
    const int base = blk ? 20 : 9;
    const float* conv_w = (const float*)d_in[base + 0];
    const float* conv_b = (const float*)d_in[base + 1];
    const float* bd     = (const float*)d_in[base + 3];
    const float* log_a  = (const float*)d_in[base + 6];
    const float* bg     = (const float*)d_in[base + 8];
    const float* bo     = (const float*)d_in[base + 10];
    const float* ng = blk ? n2_g : n1_g;
    const float* nb = blk ? n2_b : n1_b;
    float* hin  = blk ? h2 : h;
    float* hout = blk ? h  : h2;

    // LN(h) -> tmp (this is 'x' inside selective_ssm)
    layernorm_kernel<<<BL / 8, elemThreads, 0, stream>>>(hin, ng, nb, tmp, BL);
    // depthwise causal conv + SiLU -> xcv
    conv_silu_kernel<<<(BATCH * SEQ * DIM) / elemThreads, elemThreads, 0, stream>>>(
        tmp, conv_w, conv_b, xcv);
    // delta = softplus(xc @ Wd + bd)
    wmma_gemm_bf16<1><<<gGridD, gBlk, 0, stream>>>(
        xcv, DIM, p_Wd[blk], bd, nullptr, delta, DIM, BL, DIM);
    // Bm = xc @ Wb ; Cm = xc @ Wc
    wmma_gemm_bf16<0><<<gGridS, gBlk, 0, stream>>>(
        xcv, DIM, p_Wb[blk], nullptr, nullptr, Bmb, SDIM, BL, DIM);
    wmma_gemm_bf16<0><<<gGridS, gBlk, 0, stream>>>(
        xcv, DIM, p_Wc[blk], nullptr, nullptr, Cmb, SDIM, BL, DIM);
    // gate = sigmoid(x_ln @ Wg + bg)  (pre-conv input)
    wmma_gemm_bf16<2><<<gGridD, gBlk, 0, stream>>>(
        tmp, DIM, p_Wg[blk], bg, nullptr, gate, DIM, BL, DIM);
    // sequential scan (gate fused) -> yg
    scan_kernel<<<(BATCH * DIM) / 8, elemThreads, 0, stream>>>(
        delta, xcv, Bmb, Cmb, log_a, gate, yg);
    // hout = hin + (yg @ Wo + bo)   (residual fused in epilogue)
    wmma_gemm_bf16<0><<<gGridD, gBlk, 0, stream>>>(
        yg, DIM, p_Wo[blk], bo, hin, hout, DIM, BL, DIM);
  }

  // 3) heads: t = gelu(h @ w1[hh] + b1[hh]);  out[hh] = t @ w2[hh] + b2[hh]
  for (int hh = 0; hh < NHEAD; ++hh) {
    wmma_gemm_bf16<3><<<gGridH, gBlk, 0, stream>>>(
        h, DIM, p_head + (size_t)hh * DIM * DHEAD,
        head_b1 + hh * DHEAD, nullptr, thead, DHEAD, BL, DIM);
    head_dot_kernel<<<BL / 8, elemThreads, 0, stream>>>(
        thead, head_w2, head_b2, out + (size_t)hh * BL, hh);
  }
}
